// SimplifiedGCNforOOD_88484916232737
// MI455X (gfx1250) — matrix operand order repaired and checked by reading, not compiled, not run on hardware
//
#include <hip/hip_runtime.h>

#define NN 50000
#define NE 500000
#define HDIM 128
#define NH 3
#define NCLS 10
#define NG 64
#define EPSV 1e-5f
#define SLOPE 0.2f

typedef __attribute__((ext_vector_type(16))) _Float16 v16h;
typedef __attribute__((ext_vector_type(8)))  _Float16 v8h;
typedef __attribute__((ext_vector_type(8)))  float    v8f;

// ---- order-preserving float<->u32 encoding for atomicMax on floats ----
__device__ __forceinline__ unsigned enc_f32(float f) {
  int b = __float_as_int(f);
  return (b >= 0) ? (unsigned(b) | 0x80000000u) : ~unsigned(b);
}
__device__ __forceinline__ float dec_f32(unsigned u) {
  int b = (u & 0x80000000u) ? int(u & 0x7fffffffu) : int(~u);
  return __int_as_float(b);
}
__device__ __forceinline__ float lrelu(float v) { return v > 0.f ? v : SLOPE * v; }

// ---------------- conversion kernels ----------------
__global__ void k_f32_to_f16(const float* __restrict__ in, _Float16* __restrict__ out, int n) {
  int i = blockIdx.x * 256 + threadIdx.x;
  if (i < n) out[i] = (_Float16)in[i];
}
// W [K x Nout] row-major f32 -> Wt [Nout x K] row-major f16
__global__ void k_wt_f16(const float* __restrict__ W, _Float16* __restrict__ Wt, int K, int Nout) {
  int i = blockIdx.x * 256 + threadIdx.x;
  if (i < K * Nout) {
    int k = i / Nout, n = i % Nout;
    Wt[(size_t)n * K + k] = (_Float16)W[i];
  }
}

// ---------------- WMMA GEMM: C[M,Nout] (+)= A[M,128]f16 @ B[128,Nout] (B given transposed) ----
__global__ void k_gemm_wmma(const _Float16* __restrict__ A, const _Float16* __restrict__ Bt,
                            float* __restrict__ C, int Nout, int accumulate) {
  const int lane = threadIdx.x;      // one wave (32 lanes) per block
  const int l16  = lane & 15;
  const int hi   = lane >> 4;
  const int m0   = blockIdx.x * 16;
  const int n0   = blockIdx.y * 64;

  v8f c[4] = {v8f{}, v8f{}, v8f{}, v8f{}};
  if (accumulate) {
#pragma unroll
    for (int i = 0; i < 8; ++i) {
      const float* cr = C + (size_t)(m0 + hi * 8 + i) * Nout + n0 + l16;
      c[0][i] = cr[0]; c[1][i] = cr[16]; c[2][i] = cr[32]; c[3][i] = cr[48];
    }
  }
  const int arow = m0 + l16;
#pragma unroll
  for (int kb = 0; kb < 128; kb += 32) {
    // A fragment: lanes 0-15 hold K=kb..kb+15; lanes 16-31 hold K=kb+16..kb+31
    const _Float16* ap = A + (size_t)arow * 128 + kb + hi * 16;
    v8h alo = *(const v8h*)(ap);
    v8h ahi = *(const v8h*)(ap + 8);
    v16h a = __builtin_shufflevector(alo, ahi, 0,1,2,3,4,5,6,7,8,9,10,11,12,13,14,15);
#pragma unroll
    for (int nt = 0; nt < 4; ++nt) {
      // B fragment from transposed weights: 16 contiguous halves per lane
      const _Float16* bp = Bt + (size_t)(n0 + nt * 16 + l16) * 128 + kb + hi * 16;
      v16h b = *(const v16h*)(bp);
      c[nt] = __builtin_amdgcn_wmma_f32_16x16x32_f16(false, a, false, b, (short)0, c[nt],
                                                     false, false);
    }
  }
#pragma unroll
  for (int i = 0; i < 8; ++i) {
    float* cr = C + (size_t)(m0 + hi * 8 + i) * Nout + n0 + l16;
    cr[0] = c[0][i]; cr[16] = c[1][i]; cr[32] = c[2][i]; cr[48] = c[3][i];
  }
}

// ---------------- degrees ----------------
__global__ void k_deg(const int* __restrict__ ei, float* __restrict__ deg) {
  int e = blockIdx.x * 256 + threadIdx.x;
  if (e < NE) atomicAdd(&deg[ei[NE + e]], 1.0f);
}
__global__ void k_dinv(const float* __restrict__ deg, float* __restrict__ dinv) {
  int n = blockIdx.x * 256 + threadIdx.x;
  if (n < NN) dinv[n] = rsqrtf(deg[n] + 1.0f);   // +1 for self loop
}

// ---------------- GCN aggregation ----------------
__global__ void k_gcn_init(const float* __restrict__ g, const float* __restrict__ dinv,
                           float* __restrict__ acc) {
  int n = blockIdx.x, t = threadIdx.x;
  float w = dinv[n] * dinv[n];
  acc[(size_t)n * HDIM + t] = g[(size_t)n * HDIM + t] * w;   // self-loop term
}
__global__ void k_gcn_edge(const int* __restrict__ ei, const float* __restrict__ g,
                           const float* __restrict__ dinv, float* __restrict__ acc) {
  int e = blockIdx.x, t = threadIdx.x;
  int s = ei[e], d = ei[NE + e];
  float w = dinv[s] * dinv[d];
  atomicAdd(&acc[(size_t)d * HDIM + t], g[(size_t)s * HDIM + t] * w);
}

// ---------------- BatchNorm ----------------
__global__ void k_bn_stats(const float* __restrict__ h, float* __restrict__ sums,
                           float* __restrict__ sumsq) {
  int t = threadIdx.x;
  int rows_per = (NN + gridDim.x - 1) / gridDim.x;
  int r0 = blockIdx.x * rows_per;
  int r1 = r0 + rows_per; if (r1 > NN) r1 = NN;
  float s = 0.f, q = 0.f;
  for (int r = r0; r < r1; ++r) {
    float v = h[(size_t)r * HDIM + t];
    s += v; q += v * v;
  }
  atomicAdd(&sums[t], s);
  atomicAdd(&sumsq[t], q);
}
__global__ void k_bn_apply(const float* __restrict__ g, const float* __restrict__ sums,
                           const float* __restrict__ sumsq, const float* __restrict__ gamma,
                           const float* __restrict__ beta, float* __restrict__ hout,
                           _Float16* __restrict__ hf16) {
  int n = blockIdx.x, t = threadIdx.x;
  float mu  = sums[t] * (1.0f / NN);
  float var = sumsq[t] * (1.0f / NN) - mu * mu;
  float v = (g[(size_t)n * HDIM + t] - mu) * rsqrtf(var + EPSV) * gamma[t] + beta[t];
  v = v > 0.f ? v : (expf(v) - 1.f);            // ELU
  hout[(size_t)n * HDIM + t] = v;
  if (hf16) hf16[(size_t)n * HDIM + t] = (_Float16)v;
}

// ---------------- SAGE aggregation ----------------
__global__ void k_sage_edge(const int* __restrict__ ei, const float* __restrict__ h,
                            float* __restrict__ agg) {
  int e = blockIdx.x, t = threadIdx.x;
  int s = ei[e], d = ei[NE + e];
  atomicAdd(&agg[(size_t)d * HDIM + t], h[(size_t)s * HDIM + t]);
}
__global__ void k_sage_div_f16(const float* __restrict__ agg, const float* __restrict__ deg,
                               _Float16* __restrict__ out) {
  int n = blockIdx.x, t = threadIdx.x;
  float dv = fmaxf(deg[n], 1.0f);
  out[(size_t)n * HDIM + t] = (_Float16)(agg[(size_t)n * HDIM + t] / dv);
}

// ---------------- GAT ----------------
__global__ void k_gat_scores(const float* __restrict__ hw, const float* __restrict__ att_src,
                             const float* __restrict__ att_dst, float* __restrict__ a_src,
                             float* __restrict__ a_dst, unsigned* __restrict__ mmax) {
  __shared__ float s1[HDIM], s2[HDIM];
  int n = blockIdx.x, t = threadIdx.x;
  for (int h = 0; h < NH; ++h) {
    float v = hw[(size_t)n * (NH * HDIM) + h * HDIM + t];
    s1[t] = v * att_src[h * HDIM + t];
    s2[t] = v * att_dst[h * HDIM + t];
    __syncthreads();
    for (int off = 64; off > 0; off >>= 1) {
      if (t < off) { s1[t] += s1[t + off]; s2[t] += s2[t + off]; }
      __syncthreads();
    }
    if (t == 0) {
      float as = s1[0], ad = s2[0];
      a_src[(size_t)n * NH + h] = as;
      a_dst[(size_t)n * NH + h] = ad;
      mmax[(size_t)n * NH + h] = enc_f32(lrelu(as + ad));   // init with self-loop score
    }
    __syncthreads();
  }
}
__global__ void k_gat_edge_max(const int* __restrict__ ei, const float* __restrict__ a_src,
                               const float* __restrict__ a_dst, unsigned* __restrict__ mmax) {
  int i = blockIdx.x * 256 + threadIdx.x;
  if (i >= NE * NH) return;
  int e = i / NH, h = i - e * NH;
  int s = ei[e], d = ei[NE + e];
  float v = lrelu(a_src[s * NH + h] + a_dst[d * NH + h]);
  atomicMax(&mmax[d * NH + h], enc_f32(v));
}
__global__ void k_gat_denom_init(const float* __restrict__ a_src, const float* __restrict__ a_dst,
                                 const unsigned* __restrict__ mmax, float* __restrict__ denom) {
  int i = blockIdx.x * 256 + threadIdx.x;
  if (i >= NN * NH) return;
  int n = i / NH, h = i - n * NH;
  float v = lrelu(a_src[n * NH + h] + a_dst[n * NH + h]);
  denom[i] = expf(v - dec_f32(mmax[i]));                    // self-loop term
}
__global__ void k_gat_edge_denom(const int* __restrict__ ei, const float* __restrict__ a_src,
                                 const float* __restrict__ a_dst, const unsigned* __restrict__ mmax,
                                 float* __restrict__ denom) {
  int i = blockIdx.x * 256 + threadIdx.x;
  if (i >= NE * NH) return;
  int e = i / NH, h = i - e * NH;
  int s = ei[e], d = ei[NE + e];
  float v = lrelu(a_src[s * NH + h] + a_dst[d * NH + h]);
  atomicAdd(&denom[d * NH + h], expf(v - dec_f32(mmax[d * NH + h])));
}
__global__ void k_gat_out_init(const float* __restrict__ hw, const float* __restrict__ a_src,
                               const float* __restrict__ a_dst, const unsigned* __restrict__ mmax,
                               const float* __restrict__ denom, float* __restrict__ acc) {
  __shared__ float w[NH];
  int n = blockIdx.x, t = threadIdx.x;
  if (t < NH) {
    float v = lrelu(a_src[n * NH + t] + a_dst[n * NH + t]);
    w[t] = expf(v - dec_f32(mmax[n * NH + t])) / denom[n * NH + t];
  }
  __syncthreads();
  const float* hs = hw + (size_t)n * (NH * HDIM);
  acc[(size_t)n * HDIM + t] =
      (w[0] * hs[t] + w[1] * hs[HDIM + t] + w[2] * hs[2 * HDIM + t]) * (1.0f / NH);
}
__global__ void k_gat_edge_agg(const int* __restrict__ ei, const float* __restrict__ hw,
                               const float* __restrict__ a_src, const float* __restrict__ a_dst,
                               const unsigned* __restrict__ mmax, const float* __restrict__ denom,
                               float* __restrict__ acc) {
  __shared__ float w[NH];
  __shared__ int sd[2];
  int e = blockIdx.x, t = threadIdx.x;
  if (t == 0) { sd[0] = ei[e]; sd[1] = ei[NE + e]; }
  __syncthreads();
  int s = sd[0], d = sd[1];
  if (t < NH) {
    float v = lrelu(a_src[s * NH + t] + a_dst[d * NH + t]);
    w[t] = expf(v - dec_f32(mmax[d * NH + t])) / denom[d * NH + t];
  }
  __syncthreads();
  const float* hs = hw + (size_t)s * (NH * HDIM);
  float val = (w[0] * hs[t] + w[1] * hs[HDIM + t] + w[2] * hs[2 * HDIM + t]) * (1.0f / NH);
  atomicAdd(&acc[(size_t)d * HDIM + t], val);
}

// ---------------- pooling + head ----------------
__global__ void k_pool(const float* __restrict__ h, const int* __restrict__ batch,
                       float* __restrict__ poolsum, float* __restrict__ cnt) {
  int n = blockIdx.x, t = threadIdx.x;
  int g = batch[n];
  atomicAdd(&poolsum[(size_t)g * HDIM + t], h[(size_t)n * HDIM + t]);
  if (t == 0) atomicAdd(&cnt[g], 1.0f);
}
__global__ void k_head(const float* __restrict__ poolsum, const float* __restrict__ cnt,
                       const float* __restrict__ Wc, const float* __restrict__ bc,
                       float* __restrict__ out) {
  __shared__ float p[HDIM];
  int g = blockIdx.x, t = threadIdx.x;
  float c = fmaxf(cnt[g], 1.0f);
  float pv = poolsum[(size_t)g * HDIM + t] / c;
  out[(size_t)g * HDIM + t] = pv;          // pooled output
  p[t] = pv;
  __syncthreads();
  if (t < NCLS) {
    float a = bc[t];
    for (int k = 0; k < HDIM; ++k) a += p[k] * Wc[k * NCLS + t];
    out[NG * HDIM + g * NCLS + t] = a;     // logits output
  }
}

extern "C" void kernel_launch(void* const* d_in, const int* in_sizes, int n_in,
                              void* d_out, int out_size, void* d_ws, size_t ws_size,
                              hipStream_t stream) {
  (void)in_sizes; (void)n_in; (void)out_size; (void)ws_size;
  const float* x      = (const float*)d_in[0];
  const int*   ei     = (const int*)  d_in[1];
  const int*   batch  = (const int*)  d_in[2];
  const float* W1     = (const float*)d_in[3];
  const float* gamma1 = (const float*)d_in[5];
  const float* beta1  = (const float*)d_in[6];
  const float* Wl     = (const float*)d_in[7];
  const float* Wr     = (const float*)d_in[9];
  const float* gamma2 = (const float*)d_in[10];
  const float* beta2  = (const float*)d_in[11];
  const float* W3     = (const float*)d_in[12];
  const float* att_s  = (const float*)d_in[13];
  const float* att_d  = (const float*)d_in[14];
  const float* gamma3 = (const float*)d_in[16];
  const float* beta3  = (const float*)d_in[17];
  const float* Wc     = (const float*)d_in[18];
  const float* bc     = (const float*)d_in[19];
  float* out = (float*)d_out;

  // -------- workspace carve-up (256B aligned) --------
  char* ws = (char*)d_ws;
  size_t off = 0;
  auto carve = [&](size_t bytes) -> void* {
    void* p = ws + off;
    off += (bytes + 255) & ~(size_t)255;
    return p;
  };
  float*    g_out  = (float*)   carve((size_t)NN * NH * HDIM * 4); // GEMM output / GAT hW
  float*    acc    = (float*)   carve((size_t)NN * HDIM * 4);      // aggregation accumulator
  float*    hcur   = (float*)   carve((size_t)NN * HDIM * 4);      // activations f32
  _Float16* hf16a  = (_Float16*)carve((size_t)NN * HDIM * 2);      // activations f16 (GEMM A)
  _Float16* hf16b  = (_Float16*)carve((size_t)NN * HDIM * 2);      // SAGE mean-agg f16
  float*    deg    = (float*)   carve((size_t)NN * 4);
  float*    dinv   = (float*)   carve((size_t)NN * 4);
  _Float16* W1t    = (_Float16*)carve(128 * 128 * 2);
  _Float16* Wlt    = (_Float16*)carve(128 * 128 * 2);
  _Float16* Wrt    = (_Float16*)carve(128 * 128 * 2);
  _Float16* W3t    = (_Float16*)carve(384 * 128 * 2);
  float*    sums   = (float*)   carve(HDIM * 4);
  float*    sumsq  = (float*)   carve(HDIM * 4);
  float*    a_src  = (float*)   carve((size_t)NN * NH * 4);
  float*    a_dst  = (float*)   carve((size_t)NN * NH * 4);
  unsigned* mmax   = (unsigned*)carve((size_t)NN * NH * 4);
  float*    denom  = (float*)   carve((size_t)NN * NH * 4);
  float*    psum   = (float*)   carve((size_t)NG * HDIM * 4);
  float*    cnt    = (float*)   carve((size_t)NG * 4);

  // -------- one-time-per-call prep --------
  hipMemsetAsync(deg, 0, (size_t)NN * 4, stream);
  k_wt_f16<<<(128 * 128 + 255) / 256, 256, 0, stream>>>(W1, W1t, 128, 128);
  k_wt_f16<<<(128 * 128 + 255) / 256, 256, 0, stream>>>(Wl, Wlt, 128, 128);
  k_wt_f16<<<(128 * 128 + 255) / 256, 256, 0, stream>>>(Wr, Wrt, 128, 128);
  k_wt_f16<<<(128 * 384 + 255) / 256, 256, 0, stream>>>(W3, W3t, 128, 384);
  k_f32_to_f16<<<(NN * HDIM + 255) / 256, 256, 0, stream>>>(x, hf16a, NN * HDIM);
  k_deg<<<(NE + 255) / 256, 256, 0, stream>>>(ei, deg);
  k_dinv<<<(NN + 255) / 256, 256, 0, stream>>>(deg, dinv);

  // -------- layer 1: GCN --------  (bias b1 cancels under BN)
  k_gemm_wmma<<<dim3(NN / 16, HDIM / 64), 32, 0, stream>>>(hf16a, W1t, g_out, HDIM, 0);
  k_gcn_init<<<NN, HDIM, 0, stream>>>(g_out, dinv, acc);
  k_gcn_edge<<<NE, HDIM, 0, stream>>>(ei, g_out, dinv, acc);
  hipMemsetAsync(sums, 0, HDIM * 4, stream);
  hipMemsetAsync(sumsq, 0, HDIM * 4, stream);
  k_bn_stats<<<256, HDIM, 0, stream>>>(acc, sums, sumsq);
  k_bn_apply<<<NN, HDIM, 0, stream>>>(acc, sums, sumsq, gamma1, beta1, hcur, hf16a);

  // -------- layer 2: SAGE --------  (bias bl cancels under BN)
  hipMemsetAsync(acc, 0, (size_t)NN * HDIM * 4, stream);
  k_sage_edge<<<NE, HDIM, 0, stream>>>(ei, hcur, acc);
  k_sage_div_f16<<<NN, HDIM, 0, stream>>>(acc, deg, hf16b);
  k_gemm_wmma<<<dim3(NN / 16, HDIM / 64), 32, 0, stream>>>(hf16a, Wrt, g_out, HDIM, 0);
  k_gemm_wmma<<<dim3(NN / 16, HDIM / 64), 32, 0, stream>>>(hf16b, Wlt, g_out, HDIM, 1);
  hipMemsetAsync(sums, 0, HDIM * 4, stream);
  hipMemsetAsync(sumsq, 0, HDIM * 4, stream);
  k_bn_stats<<<256, HDIM, 0, stream>>>(g_out, sums, sumsq);
  k_bn_apply<<<NN, HDIM, 0, stream>>>(g_out, sums, sumsq, gamma2, beta2, hcur, hf16a);

  // -------- layer 3: GAT --------  (bias b3 cancels under BN)
  k_gemm_wmma<<<dim3(NN / 16, (NH * HDIM) / 64), 32, 0, stream>>>(hf16a, W3t, g_out, NH * HDIM, 0);
  k_gat_scores<<<NN, HDIM, 0, stream>>>(g_out, att_s, att_d, a_src, a_dst, mmax);
  k_gat_edge_max<<<(NE * NH + 255) / 256, 256, 0, stream>>>(ei, a_src, a_dst, mmax);
  k_gat_denom_init<<<(NN * NH + 255) / 256, 256, 0, stream>>>(a_src, a_dst, mmax, denom);
  k_gat_edge_denom<<<(NE * NH + 255) / 256, 256, 0, stream>>>(ei, a_src, a_dst, mmax, denom);
  k_gat_out_init<<<NN, HDIM, 0, stream>>>(g_out, a_src, a_dst, mmax, denom, acc);
  k_gat_edge_agg<<<NE, HDIM, 0, stream>>>(ei, g_out, a_src, a_dst, mmax, denom, acc);
  hipMemsetAsync(sums, 0, HDIM * 4, stream);
  hipMemsetAsync(sumsq, 0, HDIM * 4, stream);
  k_bn_stats<<<256, HDIM, 0, stream>>>(acc, sums, sumsq);
  k_bn_apply<<<NN, HDIM, 0, stream>>>(acc, sums, sumsq, gamma3, beta3, hcur, (_Float16*)nullptr);

  // -------- pool + classifier --------
  hipMemsetAsync(psum, 0, (size_t)NG * HDIM * 4, stream);
  hipMemsetAsync(cnt, 0, (size_t)NG * 4, stream);
  k_pool<<<NN, HDIM, 0, stream>>>(hcur, batch, psum, cnt);
  k_head<<<NG, HDIM, 0, stream>>>(psum, cnt, Wc, bc, out);
}
